// RoIHeads_50311246905553
// MI455X (gfx1250) — matrix operand order repaired and checked by reading, not compiled, not run on hardware
//
#include <hip/hip_runtime.h>
#include <hip/hip_bf16.h>

// ---------------- types for CDNA5 WMMA ----------------
typedef __attribute__((ext_vector_type(16))) __bf16 v16bf;
typedef __attribute__((ext_vector_type(8)))  __bf16 v8bf;
typedef __attribute__((ext_vector_type(8)))  float  v8f;
typedef unsigned short ushort_t;

// ---------------- helpers ----------------
static __device__ inline unsigned short f2bf(float f) {
    // round-to-nearest-even fp32 -> bf16 bits
    unsigned u = __float_as_uint(f);
    u += 0x7fffu + ((u >> 16) & 1u);
    return (unsigned short)(u >> 16);
}
static __device__ inline float bf2f(unsigned short h) {
    return __uint_as_float(((unsigned)h) << 16);
}
static __device__ inline v16bf frag16(const unsigned short* p0, const unsigned short* p1) {
    v8bf a = *(const v8bf*)p0;   // 16-byte LDS load (ds_load_b128)
    v8bf b = *(const v8bf*)p1;
    v16bf r;
#pragma unroll
    for (int i = 0; i < 8; i++) { r[i] = a[i]; r[i + 8] = b[i]; }
    return r;
}

// ---------------- constants ----------------
#define NROI   1024
#define CCH    256
#define FEATHW 50
#define KFEAT  12544      // 256*7*7
#define MBOX   40960      // 512*80 candidates per image
#define DW_CLIP_F 4.1351666f   // log(1000/16)

// =====================================================================
// RoI Align: 1 block per roi, 1 thread per channel.
// Writes bf16 hi/lo planes directly (A operand of GEMM-1).
// =====================================================================
__global__ __launch_bounds__(256) void roi_align_kernel(
    const float* __restrict__ feat, const float* __restrict__ props,
    unsigned short* __restrict__ X0hi, unsigned short* __restrict__ X0lo)
{
    const int roi = blockIdx.x;
    const int c   = threadIdx.x;
    const int img = roi >> 9;

    const float* p = props + (size_t)roi * 4;
    float x1 = p[0] * (1.f / 16.f), y1 = p[1] * (1.f / 16.f);
    float x2 = p[2] * (1.f / 16.f), y2 = p[3] * (1.f / 16.f);
    float rw = fmaxf(x2 - x1, 1.f), rh = fmaxf(y2 - y1, 1.f);

    int   yi0[14], xi0[14];
    float lys[14], lxs[14];
#pragma unroll
    for (int s = 0; s < 14; s++) {
        float gg = (s + 0.5f) * 0.5f;
        float ys = fminf(fmaxf(y1 + gg * (rh / 7.f), 0.f), 49.f);
        float xs = fminf(fmaxf(x1 + gg * (rw / 7.f), 0.f), 49.f);
        float yf = floorf(ys), xf = floorf(xs);
        yi0[s] = (int)yf; xi0[s] = (int)xf;
        lys[s] = ys - yf; lxs[s] = xs - xf;
    }

    const float* fc = feat + ((size_t)(img * CCH + c)) * (FEATHW * FEATHW);
    const size_t obase = (size_t)roi * KFEAT + (size_t)c * 49;

    for (int py = 0; py < 7; py++) {
        for (int px = 0; px < 7; px++) {
            float accv = 0.f;
#pragma unroll
            for (int sy = 0; sy < 2; sy++) {
                int s  = py * 2 + sy;
                int y0 = yi0[s];
                int y1i = (y0 + 1 < 49) ? y0 + 1 : 49;
                float fy = lys[s], gy = 1.f - fy;
#pragma unroll
                for (int sx = 0; sx < 2; sx++) {
                    int t  = px * 2 + sx;
                    int x0 = xi0[t];
                    int x1i = (x0 + 1 < 49) ? x0 + 1 : 49;
                    float fx = lxs[t], gx = 1.f - fx;
                    float v00 = fc[y0 * FEATHW + x0],  v01 = fc[y0 * FEATHW + x1i];
                    float v10 = fc[y1i * FEATHW + x0], v11 = fc[y1i * FEATHW + x1i];
                    accv += gy * gx * v00 + gy * fx * v01 + fy * gx * v10 + fy * fx * v11;
                }
            }
            float v = accv * 0.25f;
            unsigned short h = f2bf(v);
            X0hi[obase + py * 7 + px] = h;
            X0lo[obase + py * 7 + px] = f2bf(v - bf2f(h));
        }
    }
}

// =====================================================================
// One-time weight prep: W[K][N] f32 -> transposed bf16 planes T[N][K] hi/lo.
// Bandwidth-bound, runs once; removes all conversion VALU from GEMM loop.
// =====================================================================
__global__ __launch_bounds__(256) void convert_weight_kernel(
    const float* __restrict__ W, unsigned short* __restrict__ Thi,
    unsigned short* __restrict__ Tlo, int K, int N)
{
    int idx = blockIdx.x * 256 + threadIdx.x;
    if (idx >= K * N) return;
    int k = idx / N, n = idx - k * N;        // coalesced read along n
    float x = W[idx];
    unsigned short h = f2bf(x);
    Thi[(size_t)n * K + k] = h;
    Tlo[(size_t)n * K + k] = f2bf(x - bf2f(h));
}

// =====================================================================
// bf16x3 WMMA GEMM with pre-split operands and LDS ping-pong:
//   A planes [M][K] bf16 hi/lo, B planes pre-transposed [N][K] bf16 hi/lo.
// Block tile 128x128, 8 wave32 waves each 32x64 (2x4 WMMA tiles),
// 24 v_wmma_f32_16x16x32_bf16 per K=32 step (ah*bh + ah*bl + al*bh).
// Staging = 4 global 32B (v16bf) loads + 8 ds b128 stores per thread per step.
// out_bf16: write hi/lo planes (feeds next layer); else f32.
// =====================================================================
__global__ __launch_bounds__(256) void gemm_bf16x3(
    const unsigned short* __restrict__ Ahi, const unsigned short* __restrict__ Alo,
    const unsigned short* __restrict__ Bthi, const unsigned short* __restrict__ Btlo,
    const float* __restrict__ bias,
    float* __restrict__ Cf, unsigned short* __restrict__ Chi, unsigned short* __restrict__ Clo,
    int M, int N, int K, int do_relu, int out_bf16)
{
    __shared__ unsigned short sA[2][2][128][40];   // [buf][plane][row][k] (80B rows)
    __shared__ unsigned short sB[2][2][128][40];   // [buf][plane][n][k]

    const int tid  = threadIdx.x;
    const int lane = tid & 31;
    const int wv   = tid >> 5;
    const int wr   = wv & 3;      // rows 32*wr
    const int wc   = wv >> 2;     // cols 64*wc
    const int g    = lane >> 4;
    const int lm   = lane & 15;

    const int rowBase = blockIdx.y * 128;
    const int colBase = blockIdx.x * 128;

    // chunk q in [0,1024): q<512 -> A, else B; plane=(q>>8)&1; r=(q&255)>>1; half=q&1
    auto gload = [&](int q, int kk) -> v16bf {
        int idx = q & 255, r = idx >> 1, half = idx & 1, plane = (q >> 8) & 1;
        if (q < 512) {
            const unsigned short* s = (plane ? Alo : Ahi)
                + (size_t)(rowBase + r) * K + kk + half * 16;
            return *(const v16bf*)s;
        }
        int n = colBase + r;
        if (n < N) {
            const unsigned short* s = (plane ? Btlo : Bthi)
                + (size_t)n * K + kk + half * 16;
            return *(const v16bf*)s;
        }
        v16bf z = {}; return z;
    };
    auto lstore = [&](int buf, int q, v16bf v) {
        int idx = q & 255, r = idx >> 1, half = idx & 1, plane = (q >> 8) & 1;
        if (q < 512) { *(v8bf*)&sA[buf][plane][r][half * 16] = v.lo; *(v8bf*)&sA[buf][plane][r][half * 16 + 8] = v.hi; }
        else         { *(v8bf*)&sB[buf][plane][r][half * 16] = v.lo; *(v8bf*)&sB[buf][plane][r][half * 16 + 8] = v.hi; }
    };

    v8f acc[2][4];
#pragma unroll
    for (int i = 0; i < 2; i++)
#pragma unroll
        for (int j = 0; j < 4; j++)
#pragma unroll
            for (int e = 0; e < 8; e++) acc[i][j][e] = 0.f;

    // prologue: stage first tile into buffer 0
#pragma unroll
    for (int i = 0; i < 4; i++) {
        int q = i * 256 + tid;
        lstore(0, q, gload(q, 0));
    }
    __syncthreads();

    const int nT = K >> 5;
    for (int t = 0; t < nT; t++) {
        const int cur = t & 1;
        const bool hasNext = (t + 1) < nT;

        // issue next tile's global loads before compute (latency hiding)
        v16bf nx[4];
        if (hasNext) {
#pragma unroll
            for (int i = 0; i < 4; i++) nx[i] = gload(i * 256 + tid, (t + 1) * 32);
        }

        // ---- fragments + 24 WMMAs from buffer `cur` ----
        v16bf ahi[2], alo[2];
#pragma unroll
        for (int mt = 0; mt < 2; mt++) {
            int r = 32 * wr + 16 * mt + lm;
            ahi[mt] = frag16(&sA[cur][0][r][g * 8], &sA[cur][0][r][16 + g * 8]);
            alo[mt] = frag16(&sA[cur][1][r][g * 8], &sA[cur][1][r][16 + g * 8]);
        }
#pragma unroll
        for (int nt = 0; nt < 4; nt++) {
            int cidx = 64 * wc + 16 * nt + lm;
            v16bf bhi = frag16(&sB[cur][0][cidx][g * 8], &sB[cur][0][cidx][16 + g * 8]);
            v16bf blo = frag16(&sB[cur][1][cidx][g * 8], &sB[cur][1][cidx][16 + g * 8]);
#pragma unroll
            for (int mt = 0; mt < 2; mt++) {
                acc[mt][nt] = __builtin_amdgcn_wmma_f32_16x16x32_bf16(
                    false, ahi[mt], false, bhi, (short)0, acc[mt][nt], false, false);
                acc[mt][nt] = __builtin_amdgcn_wmma_f32_16x16x32_bf16(
                    false, ahi[mt], false, blo, (short)0, acc[mt][nt], false, false);
                acc[mt][nt] = __builtin_amdgcn_wmma_f32_16x16x32_bf16(
                    false, alo[mt], false, bhi, (short)0, acc[mt][nt], false, false);
            }
        }

        if (hasNext) {
            // store next tile into the other buffer; one barrier per iteration
#pragma unroll
            for (int i = 0; i < 4; i++) lstore(cur ^ 1, i * 256 + tid, nx[i]);
            __syncthreads();
        }
    }

    // ---- epilogue: bias (+ReLU); C/D layout row = r8 + 8*(lane/16), col = lane%16
#pragma unroll
    for (int mt = 0; mt < 2; mt++)
#pragma unroll
        for (int nt = 0; nt < 4; nt++) {
            int colT = colBase + 64 * wc + 16 * nt + lm;
            if (colT < N) {
                float bval = bias[colT];
#pragma unroll
                for (int r8 = 0; r8 < 8; r8++) {
                    int row = rowBase + 32 * wr + 16 * mt + r8 + 8 * g;
                    float v = acc[mt][nt][r8] + bval;
                    if (do_relu) v = fmaxf(v, 0.f);
                    size_t o = (size_t)row * N + colT;
                    if (out_bf16) {
                        unsigned short h = f2bf(v);
                        Chi[o] = h;
                        Clo[o] = f2bf(v - bf2f(h));
                    } else {
                        Cf[o] = v;
                    }
                }
            }
        }
}

// =====================================================================
// softmax + box decode + validity: 1 thread per roi
// =====================================================================
__global__ __launch_bounds__(256) void decode_kernel(
    const float* __restrict__ LOG, const float* __restrict__ REG,
    const float* __restrict__ props, float* __restrict__ SCN, float* __restrict__ BX)
{
    int roi = blockIdx.x * 256 + threadIdx.x;
    if (roi >= NROI) return;

    const float* lg = LOG + (size_t)roi * 81;
    float mx = lg[0];
    for (int k = 1; k < 81; k++) mx = fmaxf(mx, lg[k]);
    float sum = 0.f;
    for (int k = 0; k < 81; k++) sum += expf(lg[k] - mx);
    float inv = 1.f / sum;

    const float* p = props + (size_t)roi * 4;
    float bw = p[2] - p[0], bh = p[3] - p[1];
    float cx = p[0] + 0.5f * bw, cy = p[1] + 0.5f * bh;
    int img = roi >> 9, bi = roi & 511;
    const float NEG = -__builtin_inff();

    for (int j = 1; j < 81; j++) {
        float prob = expf(lg[j] - mx) * inv;
        const float* d = REG + (size_t)roi * 324 + j * 4;
        float dx = d[0] * 0.1f, dy = d[1] * 0.1f;
        float dw = fminf(d[2] * 0.2f, DW_CLIP_F);
        float dh = fminf(d[3] * 0.2f, DW_CLIP_F);
        float pcx = dx * bw + cx, pcy = dy * bh + cy;
        float pw = expf(dw) * bw, ph = expf(dh) * bh;
        float x1 = fminf(fmaxf(pcx - 0.5f * pw, 0.f), 800.f);
        float y1 = fminf(fmaxf(pcy - 0.5f * ph, 0.f), 800.f);
        float x2 = fminf(fmaxf(pcx + 0.5f * pw, 0.f), 800.f);
        float y2 = fminf(fmaxf(pcy + 0.5f * ph, 0.f), 800.f);
        float wv = x2 - x1, hv = y2 - y1;
        bool valid = (prob >= 0.05f) && (wv >= 1.f) && (hv >= 1.f);
        int idx = img * MBOX + bi * 80 + (j - 1);
        SCN[idx] = valid ? prob : NEG;
        float4 bb4; bb4.x = x1; bb4.y = y1; bb4.z = x2; bb4.w = y2;
        ((float4*)BX)[idx] = bb4;
    }
}

// =====================================================================
// Greedy masked NMS, 1 block per image, 100 iterations (matches lax.scan).
// out: [boxes 2*100*4][scores 2*100][labels 2*100][ok 2*100] as float.
// =====================================================================
__global__ __launch_bounds__(256) void nms_kernel(
    float* __restrict__ SCN, const float* __restrict__ BX, float* __restrict__ out)
{
    const int img = blockIdx.x;
    const int tid = threadIdx.x;
    float* sc = SCN + (size_t)img * MBOX;
    const float4* bx = (const float4*)BX + (size_t)img * MBOX;

    __shared__ float rv[256];
    __shared__ int   ri[256];
    __shared__ float cbox[4];
    __shared__ int   cok;
    const float NEG = -__builtin_inff();

    for (int it = 0; it < 100; it++) {
        float best = NEG; int bidx = 0x7fffffff;
        for (int j = tid; j < MBOX; j += 256) {
            float v = sc[j];
            if (v > best || (v == best && j < bidx)) { best = v; bidx = j; }
        }
        rv[tid] = best; ri[tid] = bidx;
        __syncthreads();
        for (int s = 128; s > 0; s >>= 1) {
            if (tid < s) {
                float v2 = rv[tid + s]; int i2 = ri[tid + s];
                if (v2 > rv[tid] || (v2 == rv[tid] && i2 < ri[tid])) { rv[tid] = v2; ri[tid] = i2; }
            }
            __syncthreads();
        }
        if (tid == 0) {
            int i = ri[0]; float s0 = rv[0];
            int ok = (s0 > NEG) ? 1 : 0;
            float4 b = bx[i];
            cbox[0] = b.x; cbox[1] = b.y; cbox[2] = b.z; cbox[3] = b.w; cok = ok;
            float* ob = out + img * 400 + it * 4;
            ob[0] = ok ? b.x : 0.f; ob[1] = ok ? b.y : 0.f;
            ob[2] = ok ? b.z : 0.f; ob[3] = ok ? b.w : 0.f;
            out[800  + img * 100 + it] = ok ? s0 : 0.f;
            out[1000 + img * 100 + it] = ok ? (float)((i % 80) + 1) : 0.f;
            out[1200 + img * 100 + it] = (float)ok;
            sc[i] = NEG;
        }
        __syncthreads();
        if (cok) {
            float bx1 = cbox[0], by1 = cbox[1], bx2 = cbox[2], by2 = cbox[3];
            float a1 = (bx2 - bx1) * (by2 - by1);
            for (int j = tid; j < MBOX; j += 256) {
                float4 q = bx[j];
                float ltx = fmaxf(bx1, q.x), lty = fmaxf(by1, q.y);
                float rbx = fminf(bx2, q.z), rby = fminf(by2, q.w);
                float iw = fmaxf(rbx - ltx, 0.f), ih = fmaxf(rby - lty, 0.f);
                float inter = iw * ih;
                float a2 = (q.z - q.x) * (q.w - q.y);
                float iou = inter / (a1 + a2 - inter + 1e-9f);
                if (iou > 0.5f) sc[j] = NEG;
            }
        }
        __syncthreads();
    }
}

// =====================================================================
extern "C" void kernel_launch(void* const* d_in, const int* in_sizes, int n_in,
                              void* d_out, int out_size, void* d_ws, size_t ws_size,
                              hipStream_t stream)
{
    (void)in_sizes; (void)n_in; (void)out_size; (void)ws_size;

    const float* features  = (const float*)d_in[0];
    const float* proposals = (const float*)d_in[1];
    const float* W1 = (const float*)d_in[2];
    const float* b1 = (const float*)d_in[3];
    const float* W2 = (const float*)d_in[4];
    const float* b2 = (const float*)d_in[5];
    const float* Wc = (const float*)d_in[6];
    const float* bc = (const float*)d_in[7];
    const float* Wb = (const float*)d_in[8];
    const float* bb = (const float*)d_in[9];
    float* out = (float*)d_out;

    // ---- carve workspace (256B-aligned slices) ----
    size_t off = 0;
    char* base = (char*)d_ws;
    auto carve = [&](size_t bytes) -> void* {
        void* p = base + off;
        off += (bytes + 255) & ~(size_t)255;
        return p;
    };
    unsigned short* X0hi = (unsigned short*)carve((size_t)NROI * KFEAT * 2);
    unsigned short* X0lo = (unsigned short*)carve((size_t)NROI * KFEAT * 2);
    unsigned short* W1hi = (unsigned short*)carve((size_t)KFEAT * 1024 * 2);
    unsigned short* W1lo = (unsigned short*)carve((size_t)KFEAT * 1024 * 2);
    unsigned short* W2hi = (unsigned short*)carve((size_t)1024 * 1024 * 2);
    unsigned short* W2lo = (unsigned short*)carve((size_t)1024 * 1024 * 2);
    unsigned short* Wchi = (unsigned short*)carve((size_t)1024 * 81 * 2);
    unsigned short* Wclo = (unsigned short*)carve((size_t)1024 * 81 * 2);
    unsigned short* Wbhi = (unsigned short*)carve((size_t)1024 * 324 * 2);
    unsigned short* Wblo = (unsigned short*)carve((size_t)1024 * 324 * 2);
    unsigned short* H1hi = (unsigned short*)carve((size_t)NROI * 1024 * 2);
    unsigned short* H1lo = (unsigned short*)carve((size_t)NROI * 1024 * 2);
    unsigned short* H2hi = (unsigned short*)carve((size_t)NROI * 1024 * 2);
    unsigned short* H2lo = (unsigned short*)carve((size_t)NROI * 1024 * 2);
    float* LOG = (float*)carve((size_t)NROI * 81 * 4);
    float* REG = (float*)carve((size_t)NROI * 324 * 4);
    float* SCN = (float*)carve((size_t)2 * MBOX * 4);
    float* BX  = (float*)carve((size_t)2 * MBOX * 16);

    // ---- stage 0: operand prep (bandwidth-bound, one pass) ----
    roi_align_kernel<<<NROI, 256, 0, stream>>>(features, proposals, X0hi, X0lo);
    convert_weight_kernel<<<(KFEAT * 1024 + 255) / 256, 256, 0, stream>>>(W1, W1hi, W1lo, KFEAT, 1024);
    convert_weight_kernel<<<(1024 * 1024 + 255) / 256, 256, 0, stream>>>(W2, W2hi, W2lo, 1024, 1024);
    convert_weight_kernel<<<(1024 * 81 + 255) / 256, 256, 0, stream>>>(Wc, Wchi, Wclo, 1024, 81);
    convert_weight_kernel<<<(1024 * 324 + 255) / 256, 256, 0, stream>>>(Wb, Wbhi, Wblo, 1024, 324);

    // ---- stage 1: WMMA GEMM chain ----
    gemm_bf16x3<<<dim3(8, 8), 256, 0, stream>>>(X0hi, X0lo, W1hi, W1lo, b1,
        nullptr, H1hi, H1lo, NROI, 1024, KFEAT, 1, 1);
    gemm_bf16x3<<<dim3(8, 8), 256, 0, stream>>>(H1hi, H1lo, W2hi, W2lo, b2,
        nullptr, H2hi, H2lo, NROI, 1024, 1024, 1, 1);
    gemm_bf16x3<<<dim3(1, 8), 256, 0, stream>>>(H2hi, H2lo, Wchi, Wclo, bc,
        LOG, nullptr, nullptr, NROI, 81, 1024, 0, 0);
    gemm_bf16x3<<<dim3(3, 8), 256, 0, stream>>>(H2hi, H2lo, Wbhi, Wblo, bb,
        REG, nullptr, nullptr, NROI, 324, 1024, 0, 0);

    // ---- stage 2: detection post-processing ----
    decode_kernel<<<4, 256, 0, stream>>>(LOG, REG, proposals, SCN, BX);
    nms_kernel<<<2, 256, 0, stream>>>(SCN, BX, out);
}
